// LSTM_39247411151126
// MI455X (gfx1250) — compile-verified, matrix operand order checked by reading, not run on hardware
//
#include <hip/hip_runtime.h>

// ---------------------------------------------------------------------------
// Fused LSTM cell for MI455X (gfx1250, wave32, WMMA).
//   gates = [x|h] @ W   (1024x4096 @ 4096x8192, fp32 in, bf16 WMMA, fp32 acc)
//   c' = c*sig(f) + sig(i)*tanh(g);  h' = sig(o)*tanh(c')
// Each block: 128 rows x 32 cols of ALL FOUR gates (A tile shared), gating
// fused in the epilogue. 512 threads = 16 wave32 waves.
// Double-buffered LDS: ONE barrier per K-step; staging VALU (pk-bf16 cvt)
// co-executes with WMMA; global loads lead by two iterations.
// ---------------------------------------------------------------------------

typedef __attribute__((ext_vector_type(16))) __bf16 v16bf;
typedef __attribute__((ext_vector_type(4)))  __bf16 v4bf;
typedef __attribute__((ext_vector_type(2)))  __bf16 v2bf;
typedef __attribute__((ext_vector_type(8)))  float  v8f;

#define BATCH   1024
#define DHSUM   4096   // D + H
#define HID     2048
#define WSTRIDE 8192   // 4*HID
#define BM      128
#define BN      32     // per gate
#define BK      32
#define KTILES  (DHSUM / BK)   // 128
#define NTHR    512
#define LDA     40     // padded u16 stride (bank spread, 8B-aligned chunks)

// Packed f32 -> bf16x2; lowers to v_cvt_pk_bf16_f32 (verified in disasm).
__device__ __forceinline__ unsigned pack2(float lo, float hi) {
  v2bf v;
  v.x = (__bf16)lo;
  v.y = (__bf16)hi;
  return __builtin_bit_cast(unsigned, v);
}
// Cheap transcendentals on v_exp_f32; correct limits at +/-inf.
__device__ __forceinline__ float sigm(float x) { return 1.0f / (1.0f + __expf(-x)); }
__device__ __forceinline__ float tanh_fast(float x) {
  return 2.0f / (1.0f + __expf(-2.0f * x)) - 1.0f;
}

// Fragment loader for both A (16x32, [row][k]) and B (32x16, stored [n][k]).
// ISA layout: lanes 0-15 -> K 0-7 (v0-3) and K 16-23 (v4-7);
//             lanes 16-31 -> K 8-15 and K 24-31.  Two 8-u16 chunks -> 4x b64.
__device__ __forceinline__ v16bf frag_load(const unsigned short* row, int half) {
  union { v16bf v; v4bf q[4]; } f;
  const v4bf* p0 = (const v4bf*)(row + 8 * half);        // K = 8*half .. +7
  const v4bf* p1 = (const v4bf*)(row + 16 + 8 * half);   // K = 16+8*half .. +7
  f.q[0] = p0[0]; f.q[1] = p0[1];
  f.q[2] = p1[0]; f.q[3] = p1[1];
  return f.v;
}

struct Frags { v16bf a, f, i, g, o; };

__global__ __launch_bounds__(NTHR)
void lstm_wmma_kernel(const float* __restrict__ x, const float* __restrict__ h,
                      const float* __restrict__ c, const float* __restrict__ W,
                      float* __restrict__ out) {
  __shared__ unsigned short lds_a[2][BM * LDA];        // [buf][row][k]   2x10240 B
  __shared__ unsigned short lds_b[2][4 * BN * LDA];    // [buf][g][n][k]  2x10240 B

  const int tid = threadIdx.x;
  const int bn  = blockIdx.x;    // N tile: cols bn*32 within each gate
  const int bm  = blockIdx.y;    // M tile: rows bm*128

  // ---- global->LDS staging mapping
  const int ar  = tid >> 2;          // A: row 0..127
  const int acq = tid & 3;           // A: 8-float column group
  const int bg  = tid >> 7;          // B: gate 0..3
  const int bt  = tid & 127;
  const int bkp = bt >> 3;           // B: K pair 0..15
  const int bnq = (bt & 7) * 4;      // B: n offset, 4 cols

  // ---- wave/fragment mapping
  const int lane = tid & 31;
  const int wv   = tid >> 5;         // 0..15
  const int msub = wv & 7;           // 16-row sub-tile
  const int nsub = wv >> 3;          // 16-col sub-tile
  const int half = lane >> 4;
  const int l    = lane & 15;

  v8f accF = {}, accI = {}, accG = {}, accO = {};
  float4 ra0, ra1, rb0, rb1;

  // Hoisted per-thread base pointers (loop adds only kt*BK).
  const float* aX = x + (size_t)(bm * BM + ar) * 2048 + acq * 8;
  const float* aH = h + (size_t)(bm * BM + ar) * 2048 + acq * 8;
  const float* bW = W + (size_t)(2 * bkp) * WSTRIDE
                      + (size_t)bg * HID + bn * BN + bnq;

  auto gload = [&](int kt) {
    const float* asrc = (kt < KTILES / 2) ? (aX + kt * BK)
                                          : (aH + (kt - KTILES / 2) * BK);
    ra0 = ((const float4*)asrc)[0];
    ra1 = ((const float4*)asrc)[1];
    const float* bsrc = bW + (size_t)(kt * BK) * WSTRIDE;
    rb0 = ((const float4*)bsrc)[0];             // K row 2*bkp
    rb1 = ((const float4*)(bsrc + WSTRIDE))[0]; // K row 2*bkp+1
  };

  auto stage = [&](int buf) {
    // A tile: bf16 row-major, contiguous K
    unsigned short* pa = &lds_a[buf][ar * LDA + acq * 8];
    ((uint2*)pa)[0]       = make_uint2(pack2(ra0.x, ra0.y), pack2(ra0.z, ra0.w));
    ((uint2*)(pa + 4))[0] = make_uint2(pack2(ra1.x, ra1.y), pack2(ra1.z, ra1.w));
    // B tile: transpose to [n][k]; u16[j]==K=j since pair (even,odd) at 2*bkp
    const float* w0 = &rb0.x;
    const float* w1 = &rb1.x;
#pragma unroll
    for (int j = 0; j < 4; ++j) {
      unsigned short* q = &lds_b[buf][(bg * BN + bnq + j) * LDA + 2 * bkp];
      *(unsigned*)q = pack2(w0[j], w1[j]);
    }
  };

  auto load_frags = [&](int buf) {
    Frags fr;
    fr.a = frag_load(&lds_a[buf][(msub * 16 + l) * LDA], half);
    fr.f = frag_load(&lds_b[buf][(0 * BN + nsub * 16 + l) * LDA], half);
    fr.i = frag_load(&lds_b[buf][(1 * BN + nsub * 16 + l) * LDA], half);
    fr.g = frag_load(&lds_b[buf][(2 * BN + nsub * 16 + l) * LDA], half);
    fr.o = frag_load(&lds_b[buf][(3 * BN + nsub * 16 + l) * LDA], half);
    return fr;
  };

  auto mma = [&](const Frags& fr) {
    accF = __builtin_amdgcn_wmma_f32_16x16x32_bf16(false, fr.a, false, fr.f,
                                                   (short)0, accF, false, false);
    accI = __builtin_amdgcn_wmma_f32_16x16x32_bf16(false, fr.a, false, fr.i,
                                                   (short)0, accI, false, false);
    accG = __builtin_amdgcn_wmma_f32_16x16x32_bf16(false, fr.a, false, fr.g,
                                                   (short)0, accG, false, false);
    accO = __builtin_amdgcn_wmma_f32_16x16x32_bf16(false, fr.a, false, fr.o,
                                                   (short)0, accO, false, false);
  };

  // ---- pipeline prologue
  gload(0);
  stage(0);
  gload(1);

  // ---- main loop: one barrier per K-step (double-buffered LDS)
  for (int kt = 0; kt < KTILES; ++kt) {
    __syncthreads();                       // buf[kt&1] staged; buf[(kt+1)&1] free
    Frags fr = load_frags(kt & 1);         // issue DS reads first
    if (kt + 1 < KTILES) {
      stage((kt + 1) & 1);                 // cvt VALU + DS stores fill load latency
      if (kt + 2 < KTILES) gload(kt + 2);  // global loads lead by 2 iterations
    }
    mma(fr);                               // WMMAs co-execute with staging VALU
  }

  // ---- fused LSTM gating epilogue (C/D layout: M = e + 8*half, N = l)
  const int row0 = bm * BM + msub * 16 + 8 * half;
  const int col  = bn * BN + nsub * 16 + l;
#pragma unroll
  for (int e = 0; e < 8; ++e) {
    const size_t idx = (size_t)(row0 + e) * HID + col;
    const float cn = c[idx] * sigm(accF[e]) + sigm(accI[e]) * tanh_fast(accG[e]);
    out[idx] = sigm(accO[e]) * tanh_fast(cn);
  }
}

extern "C" void kernel_launch(void* const* d_in, const int* in_sizes, int n_in,
                              void* d_out, int out_size, void* d_ws, size_t ws_size,
                              hipStream_t stream) {
  (void)in_sizes; (void)n_in; (void)out_size; (void)d_ws; (void)ws_size;
  const float* x = (const float*)d_in[0];
  const float* h = (const float*)d_in[1];
  const float* c = (const float*)d_in[2];
  const float* W = (const float*)d_in[3];
  float* out = (float*)d_out;
  dim3 grid(HID / BN, BATCH / BM);   // (64, 8) = 512 blocks
  lstm_wmma_kernel<<<grid, NTHR, 0, stream>>>(x, h, c, W, out);
}
